// GNNDenoiser_64183991271892
// MI455X (gfx1250) — matrix operand order repaired and checked by reading, not compiled, not run on hardware
//
#include <hip/hip_runtime.h>
#include <hip/hip_bf16.h>
#include <cstdint>

// ---------------------------------------------------------------------------
// GNN denoiser for MI455X (gfx1250, wave32).
// All GEMMs run on v_wmma_f32_16x16x32_bf16. Block = 256 threads = 8 waves;
// block tile = 64(M) x 128(N); wave w: m-tile = w&3, n-tiles = (w>>2)*4..+3.
// Software-pipelined staging: fetch(k+2: global->VGPR) and commit(k+1:
// VGPR->LDS) are placed AFTER the WMMAs of chunk k, so global-load latency is
// hidden by matrix math inside each wave (not just by occupancy). LDS ping-
// pong, one barrier per K-chunk, all LDS fragment traffic via 16B b128 ops.
// ---------------------------------------------------------------------------

#define H      128
#define CODE   1024
#define NLAYER 4

typedef __attribute__((ext_vector_type(16))) __bf16    v16bf;
typedef __attribute__((ext_vector_type(8)))  float     v8f;
typedef __attribute__((ext_vector_type(4)))  unsigned  u32x4;

union Frag { unsigned u[8]; u32x4 q[2]; v16bf v; };

__device__ inline unsigned short f2bf(float x) {
  unsigned u = __float_as_uint(x);
  u += 0x7FFFu + ((u >> 16) & 1u);  // round-to-nearest-even
  return (unsigned short)(u >> 16);
}
__device__ inline unsigned pack2(float lo, float hi) {
  return (unsigned)f2bf(lo) | ((unsigned)f2bf(hi) << 16);
}
__device__ inline float silu_f(float x) { return x / (1.0f + __expf(-x)); }

__device__ inline v8f vzero8() {
  v8f z;
#pragma unroll
  for (int i = 0; i < 8; ++i) z[i] = 0.f;
  return z;
}

__device__ inline v8f wmma_bf16(v16bf a, v16bf b, v8f c) {
  return __builtin_amdgcn_wmma_f32_16x16x32_bf16(false, a, false, b, (short)0, c,
                                                 false, false);
}

// A fragment (16x32 bf16). ISA 7.12.2: lanes 0-15 = rows with K pairs
// {0..3, 8..11}; lanes 16-31 same rows, pairs {4..7, 12..15}.
__device__ inline v16bf readA_lds(const unsigned* lds, int rowStride, int mt,
                                  int kOff, int lane) {
  Frag f;
  int row = mt * 16 + (lane & 15);
  const u32x4* b4 = (const u32x4*)(lds + row * rowStride + kOff);
  int i0 = (lane & 16) ? 1 : 0;
  f.q[0] = b4[i0];
  f.q[1] = b4[i0 + 2];
  return f.v;
}

// B fragment (32x16 bf16). LDS holds B transposed [n][k-pairs], 16 u32/row.
__device__ inline v16bf readB_lds(const unsigned* lds, int nt, int lane) {
  Frag f;
  const u32x4* b4 =
      (const u32x4*)(lds + (nt * 16 + (lane & 15)) * 16 + ((lane & 16) ? 4 : 0));
  f.q[0] = b4[0];
  f.q[1] = b4[2];
  return f.v;
}

// ---------------------------------------------------------------------------
// Weight prep: dst[n*Kpad+k] = bf16(src[k*N+n]), zero-padded to Kpad.
// ---------------------------------------------------------------------------
__global__ void wprep_kernel(const float* __restrict__ src,
                             unsigned short* __restrict__ dst, int K, int N,
                             int Kpad) {
  int total = N * Kpad;
  for (int i = blockIdx.x * blockDim.x + threadIdx.x; i < total;
       i += gridDim.x * blockDim.x) {
    int n = i / Kpad, k = i - n * Kpad;
    float v = (k < K) ? src[(size_t)k * N + n] : 0.f;
    dst[i] = f2bf(v);
  }
}

__global__ void zero_kernel(float* p, int n) {
  for (int i = blockIdx.x * blockDim.x + threadIdx.x; i < n;
       i += gridDim.x * blockDim.x)
    p[i] = 0.f;
}

__global__ void edge_prep_kernel(const float* __restrict__ coords,
                                 const int* __restrict__ ei, int E,
                                 float* __restrict__ dist,
                                 float* __restrict__ deg) {
  int e = blockIdx.x * blockDim.x + threadIdx.x;
  if (e >= E) return;
  int r = ei[e], c = ei[E + e];
  float dx = coords[3 * c + 0] - coords[3 * r + 0];
  float dy = coords[3 * c + 1] - coords[3 * r + 1];
  float dz = coords[3 * c + 2] - coords[3 * r + 2];
  dist[e] = sqrtf(dx * dx + dy * dy + dz * dz);
  atomicAdd(deg + c, 1.0f);
}

__global__ void deg_inv_kernel(float* deg, int n) {
  int i = blockIdx.x * blockDim.x + threadIdx.x;
  if (i < n) deg[i] = 1.0f / fmaxf(deg[i], 1.0f);
}

// ---------------------------------------------------------------------------
// LayerNorm: one wave32 per node (128 feats, 4/lane), writes f32 + bf16 copies
// ---------------------------------------------------------------------------
__global__ __launch_bounds__(256) void ln_kernel(
    const float* __restrict__ h, const float* __restrict__ g,
    const float* __restrict__ b, float* __restrict__ hn,
    unsigned short* __restrict__ hnb) {
  int node = blockIdx.x * 8 + (threadIdx.x >> 5);
  int lane = threadIdx.x & 31;
  float4 x4 = ((const float4*)(h + (size_t)node * H))[lane];
  float s = x4.x + x4.y + x4.z + x4.w;
#pragma unroll
  for (int m = 16; m >= 1; m >>= 1) s += __shfl_xor(s, m, 32);
  float mu = s * (1.0f / H);
  float d0 = x4.x - mu, d1 = x4.y - mu, d2 = x4.z - mu, d3 = x4.w - mu;
  float v = d0 * d0 + d1 * d1 + d2 * d2 + d3 * d3;
#pragma unroll
  for (int m = 16; m >= 1; m >>= 1) v += __shfl_xor(v, m, 32);
  float rs = rsqrtf(v * (1.0f / H) + 1e-5f);
  float4 g4 = ((const float4*)g)[lane];
  float4 b4 = ((const float4*)b)[lane];
  float o0 = d0 * rs * g4.x + b4.x, o1 = d1 * rs * g4.y + b4.y;
  float o2 = d2 * rs * g4.z + b4.z, o3 = d3 * rs * g4.w + b4.w;
  ((float4*)(hn + (size_t)node * H))[lane] = make_float4(o0, o1, o2, o3);
  unsigned* hb = (unsigned*)(hnb + (size_t)node * H) + lane * 2;
  hb[0] = pack2(o0, o1);
  hb[1] = pack2(o2, o3);
}

// ---------------------------------------------------------------------------
// Input projection: h = y @ W_in + b_in   (M=16384, K=1024, N=128)
// ---------------------------------------------------------------------------
__global__ __launch_bounds__(256) void gemm_in_kernel(
    const float* __restrict__ y, const unsigned short* __restrict__ Wt,
    const float* __restrict__ bias, float* __restrict__ h) {
  __shared__ unsigned ldsA[2][64 * 16];
  __shared__ unsigned ldsB[2][128 * 16];
  int t = threadIdx.x, lane = t & 31, w = t >> 5;
  int mt = w & 3, nh = w >> 2;
  size_t m0 = (size_t)blockIdx.x * 64;
  const int KC = CODE / 32;
  int r = t >> 2, q = t & 3;
  int nB = t >> 1, halfB = t & 1;

  float4 pa0, pa1;   // prefetched A (raw f32)
  u32x4 pb0, pb1;    // prefetched B (bf16 rows)

  auto fetchA = [&](int kc) {
    const float4* s = (const float4*)(y + (m0 + r) * CODE + kc * 32 + q * 8);
    pa0 = s[0];
    pa1 = s[1];
  };
  auto commitA = [&](unsigned* La) {
    u32x4 vv;
    vv[0] = pack2(pa0.x, pa0.y); vv[1] = pack2(pa0.z, pa0.w);
    vv[2] = pack2(pa1.x, pa1.y); vv[3] = pack2(pa1.z, pa1.w);
    *(u32x4*)(La + r * 16 + q * 4) = vv;
  };
  auto fetchB = [&](int kc) {
    const u32x4* s =
        (const u32x4*)(Wt + (size_t)nB * CODE + kc * 32 + halfB * 16);
    pb0 = s[0];
    pb1 = s[1];
  };
  auto commitB = [&](unsigned* Lb) {
    u32x4* d = (u32x4*)(Lb + nB * 16 + halfB * 8);
    d[0] = pb0;
    d[1] = pb1;
  };

  v8f acc[4];
#pragma unroll
  for (int i = 0; i < 4; ++i) acc[i] = vzero8();

  fetchA(0); fetchB(0);
  commitA(ldsA[0]); commitB(ldsB[0]);
  fetchA(1); fetchB(1);
  __syncthreads();
  for (int kc = 0; kc < KC; ++kc) {
    v16bf a = readA_lds(ldsA[kc & 1], 16, mt, 0, lane);
#pragma unroll
    for (int tt = 0; tt < 4; ++tt) {
      v16bf bb = readB_lds(ldsB[kc & 1], nh * 4 + tt, lane);
      acc[tt] = wmma_bf16(a, bb, acc[tt]);
    }
    if (kc + 1 < KC) { commitA(ldsA[(kc + 1) & 1]); commitB(ldsB[(kc + 1) & 1]); }
    if (kc + 2 < KC) { fetchA(kc + 2); fetchB(kc + 2); }
    __syncthreads();
  }
  int mbase = mt * 16 + ((lane & 16) ? 8 : 0);
#pragma unroll
  for (int tt = 0; tt < 4; ++tt) {
    int n = (nh * 4 + tt) * 16 + (lane & 15);
    float bn = bias[n];
#pragma unroll
    for (int rr = 0; rr < 8; ++rr)
      h[(m0 + mbase + rr) * H + n] = acc[tt][rr] + bn;
  }
}

// ---------------------------------------------------------------------------
// Edge MLP + scatter: m = silu(silu([hn[row],hn[col],dist] @ W1 + b1) @ W2 + b2)
// then agg[col] += m (f32 atomics). 64 edges/block, K1 padded 257->288.
// ---------------------------------------------------------------------------
__global__ __launch_bounds__(256) void edge_kernel(
    const int* __restrict__ ei, int E, const unsigned short* __restrict__ hnb,
    const float* __restrict__ dist, const unsigned short* __restrict__ W1t,
    const float* __restrict__ b1, const unsigned short* __restrict__ W2t,
    const float* __restrict__ b2, float* __restrict__ agg) {
  __shared__ unsigned ldsA[2][64 * 16];
  __shared__ unsigned ldsB[2][128 * 16];
  __shared__ unsigned short ldsM[64 * H];
  __shared__ int s_row[64];
  __shared__ int s_col[64];
  __shared__ float s_dist[64];
  int t = threadIdx.x, lane = t & 31, w = t >> 5;
  int mt = w & 3, nh = w >> 2;
  int e0 = blockIdx.x * 64;
  int r = t >> 2, q = t & 3;
  int nB = t >> 1, halfB = t & 1;
  if (t < 64) {
    int e = e0 + t;
    s_row[t] = (e < E) ? ei[e] : -1;
  } else if (t < 128) {
    int i = t - 64, e = e0 + i;
    s_col[i] = (e < E) ? ei[E + e] : -1;
  } else if (t < 192) {
    int i = t - 128, e = e0 + i;
    s_dist[i] = (e < E) ? dist[e] : 0.f;
  }
  __syncthreads();

  u32x4 rawA;
  u32x4 pb0, pb1;

  auto fetchA = [&](int kc) {
    if (kc < 8) {
      int node = (kc < 4) ? s_row[r] : s_col[r];
      if (node >= 0) {
        rawA = *(const u32x4*)(hnb + (size_t)node * H + (kc & 3) * 32 + q * 8);
      } else {
        u32x4 z; z[0] = z[1] = z[2] = z[3] = 0u;
        rawA = z;
      }
    }  // kc==8: composed at commit from s_dist
  };
  auto commitA = [&](int kc, unsigned* La) {
    u32x4* d = (u32x4*)(La + r * 16 + q * 4);
    if (kc < 8) {
      *d = rawA;
    } else {  // k=256 holds dist; 257..287 zero
      u32x4 z;
      z[0] = (q == 0) ? (unsigned)f2bf(s_dist[r]) : 0u;
      z[1] = 0u; z[2] = 0u; z[3] = 0u;
      *d = z;
    }
  };
  auto fetchB = [&](const unsigned short* Wt, int Kpad, int kc) {
    const u32x4* s =
        (const u32x4*)(Wt + (size_t)nB * Kpad + kc * 32 + halfB * 16);
    pb0 = s[0];
    pb1 = s[1];
  };
  auto commitB = [&](unsigned* Lb) {
    u32x4* d = (u32x4*)(Lb + nB * 16 + halfB * 8);
    d[0] = pb0;
    d[1] = pb1;
  };

  v8f acc[4];
#pragma unroll
  for (int i = 0; i < 4; ++i) acc[i] = vzero8();

  // GEMM1: K = 288 (9 chunks): hn[row] | hn[col] | dist-pad
  fetchA(0); fetchB(W1t, 288, 0);
  commitA(0, ldsA[0]); commitB(ldsB[0]);
  fetchA(1); fetchB(W1t, 288, 1);
  __syncthreads();
  for (int kc = 0; kc < 9; ++kc) {
    v16bf a = readA_lds(ldsA[kc & 1], 16, mt, 0, lane);
#pragma unroll
    for (int tt = 0; tt < 4; ++tt) {
      v16bf bb = readB_lds(ldsB[kc & 1], nh * 4 + tt, lane);
      acc[tt] = wmma_bf16(a, bb, acc[tt]);
    }
    if (kc + 1 < 9) { commitA(kc + 1, ldsA[(kc + 1) & 1]); commitB(ldsB[(kc + 1) & 1]); }
    if (kc + 2 < 9) {
      fetchA(kc + 2);
      fetchB(W1t, 288, kc + 2);
    } else if (kc + 2 == 9) {
      fetchB(W2t, H, 0);  // chain: prefetch W2 chunk 0 during last GEMM1 iter
    }
    __syncthreads();
  }
  // m1 = silu(acc + b1) -> LDS (bf16), reset acc
  int mbase = mt * 16 + ((lane & 16) ? 8 : 0);
#pragma unroll
  for (int tt = 0; tt < 4; ++tt) {
    int n = (nh * 4 + tt) * 16 + (lane & 15);
    float bn = b1[n];
#pragma unroll
    for (int rr = 0; rr < 8; ++rr)
      ldsM[(mbase + rr) * H + n] = f2bf(silu_f(acc[tt][rr] + bn));
    acc[tt] = vzero8();
  }
  commitB(ldsB[0]);       // W2 chunk 0 (ldsB[0] free after final GEMM1 barrier)
  fetchB(W2t, H, 1);
  __syncthreads();
  // GEMM2: K = 128 (A = m1 from LDS; single-written, no ping-pong needed)
  const unsigned* ldsMu = (const unsigned*)ldsM;
  for (int kc = 0; kc < 4; ++kc) {
    v16bf a = readA_lds(ldsMu, 64, mt, kc * 16, lane);
#pragma unroll
    for (int tt = 0; tt < 4; ++tt) {
      v16bf bb = readB_lds(ldsB[kc & 1], nh * 4 + tt, lane);
      acc[tt] = wmma_bf16(a, bb, acc[tt]);
    }
    if (kc + 1 < 4) commitB(ldsB[(kc + 1) & 1]);
    if (kc + 2 < 4) fetchB(W2t, H, kc + 2);
    __syncthreads();
  }
  // scatter-add directly from the WMMA accumulator lane layout
#pragma unroll
  for (int tt = 0; tt < 4; ++tt) {
    int n = (nh * 4 + tt) * 16 + (lane & 15);
    float bn = b2[n];
#pragma unroll
    for (int rr = 0; rr < 8; ++rr) {
      int cn = s_col[mbase + rr];
      if (cn >= 0) atomicAdd(agg + (size_t)cn * H + n, silu_f(acc[tt][rr] + bn));
    }
  }
}

// ---------------------------------------------------------------------------
// Node MLP: h = hn + (silu([hn, agg/deg] @ nW1 + nb1) @ nW2 + nb2)
// ---------------------------------------------------------------------------
__global__ __launch_bounds__(256) void node_kernel(
    const float* __restrict__ hn, const unsigned short* __restrict__ hnb,
    const float* __restrict__ agg, const float* __restrict__ invdeg,
    const unsigned short* __restrict__ W1t, const float* __restrict__ b1,
    const unsigned short* __restrict__ W2t, const float* __restrict__ b2,
    float* __restrict__ h) {
  __shared__ unsigned ldsA[2][64 * 16];
  __shared__ unsigned ldsB[2][128 * 16];
  __shared__ unsigned short ldsM[64 * H];
  int t = threadIdx.x, lane = t & 31, w = t >> 5;
  int mt = w & 3, nh = w >> 2;
  size_t m0 = (size_t)blockIdx.x * 64;
  int r = t >> 2, q = t & 3;
  int nB = t >> 1, halfB = t & 1;
  size_t nodeR = m0 + r;

  u32x4 rawA;        // kc<4: bf16 copy
  float4 ra0, ra1;   // kc>=4: raw agg f32
  float rwd;
  u32x4 pb0, pb1;

  auto fetchA = [&](int kc) {
    if (kc < 4) {
      rawA = *(const u32x4*)(hnb + nodeR * H + (kc & 3) * 32 + q * 8);
    } else {
      rwd = invdeg[nodeR];
      const float4* s = (const float4*)(agg + nodeR * H + (kc - 4) * 32 + q * 8);
      ra0 = s[0];
      ra1 = s[1];
    }
  };
  auto commitA = [&](int kc, unsigned* La) {
    u32x4* d = (u32x4*)(La + r * 16 + q * 4);
    if (kc < 4) {
      *d = rawA;
    } else {
      u32x4 vv;
      vv[0] = pack2(ra0.x * rwd, ra0.y * rwd);
      vv[1] = pack2(ra0.z * rwd, ra0.w * rwd);
      vv[2] = pack2(ra1.x * rwd, ra1.y * rwd);
      vv[3] = pack2(ra1.z * rwd, ra1.w * rwd);
      *d = vv;
    }
  };
  auto fetchB = [&](const unsigned short* Wt, int Kpad, int kc) {
    const u32x4* s =
        (const u32x4*)(Wt + (size_t)nB * Kpad + kc * 32 + halfB * 16);
    pb0 = s[0];
    pb1 = s[1];
  };
  auto commitB = [&](unsigned* Lb) {
    u32x4* d = (u32x4*)(Lb + nB * 16 + halfB * 8);
    d[0] = pb0;
    d[1] = pb1;
  };

  v8f acc[4];
#pragma unroll
  for (int i = 0; i < 4; ++i) acc[i] = vzero8();

  // GEMM1: K = 256 : [hn (bf16 copy) | agg * invdeg]
  fetchA(0); fetchB(W1t, 2 * H, 0);
  commitA(0, ldsA[0]); commitB(ldsB[0]);
  fetchA(1); fetchB(W1t, 2 * H, 1);
  __syncthreads();
  for (int kc = 0; kc < 8; ++kc) {
    v16bf a = readA_lds(ldsA[kc & 1], 16, mt, 0, lane);
#pragma unroll
    for (int tt = 0; tt < 4; ++tt) {
      v16bf bb = readB_lds(ldsB[kc & 1], nh * 4 + tt, lane);
      acc[tt] = wmma_bf16(a, bb, acc[tt]);
    }
    if (kc + 1 < 8) { commitA(kc + 1, ldsA[(kc + 1) & 1]); commitB(ldsB[(kc + 1) & 1]); }
    if (kc + 2 < 8) {
      fetchA(kc + 2);
      fetchB(W1t, 2 * H, kc + 2);
    } else if (kc + 2 == 8) {
      fetchB(W2t, H, 0);  // chain: prefetch W2 chunk 0
    }
    __syncthreads();
  }
  int mbase = mt * 16 + ((lane & 16) ? 8 : 0);
#pragma unroll
  for (int tt = 0; tt < 4; ++tt) {
    int n = (nh * 4 + tt) * 16 + (lane & 15);
    float bn = b1[n];
#pragma unroll
    for (int rr = 0; rr < 8; ++rr)
      ldsM[(mbase + rr) * H + n] = f2bf(silu_f(acc[tt][rr] + bn));
    acc[tt] = vzero8();
  }
  commitB(ldsB[0]);
  fetchB(W2t, H, 1);
  __syncthreads();
  const unsigned* ldsMu = (const unsigned*)ldsM;
  for (int kc = 0; kc < 4; ++kc) {
    v16bf a = readA_lds(ldsMu, 64, mt, kc * 16, lane);
#pragma unroll
    for (int tt = 0; tt < 4; ++tt) {
      v16bf bb = readB_lds(ldsB[kc & 1], nh * 4 + tt, lane);
      acc[tt] = wmma_bf16(a, bb, acc[tt]);
    }
    if (kc + 1 < 4) commitB(ldsB[(kc + 1) & 1]);
    if (kc + 2 < 4) fetchB(W2t, H, kc + 2);
    __syncthreads();
  }
#pragma unroll
  for (int tt = 0; tt < 4; ++tt) {
    int n = (nh * 4 + tt) * 16 + (lane & 15);
    float bn = b2[n];
#pragma unroll
    for (int rr = 0; rr < 8; ++rr) {
      size_t node = m0 + mbase + rr;
      h[node * H + n] = hn[node * H + n] + acc[tt][rr] + bn;
    }
  }
}

// ---------------------------------------------------------------------------
// Output projection: out = h @ W_out + b_out   (M=16384, K=128, N=1024)
// ---------------------------------------------------------------------------
__global__ __launch_bounds__(256) void gemm_out_kernel(
    const float* __restrict__ h, const unsigned short* __restrict__ Wt,
    const float* __restrict__ bias, float* __restrict__ out) {
  __shared__ unsigned ldsA[2][64 * 16];
  __shared__ unsigned ldsB[2][128 * 16];
  int t = threadIdx.x, lane = t & 31, w = t >> 5;
  int mt = w & 3, nh = w >> 2;
  size_t m0 = (size_t)blockIdx.x * 64;
  int n0 = blockIdx.y * 128;
  int r = t >> 2, q = t & 3;
  int nB = t >> 1, halfB = t & 1;

  float4 pa0, pa1;
  u32x4 pb0, pb1;

  auto fetchA = [&](int kc) {
    const float4* s = (const float4*)(h + (m0 + r) * H + kc * 32 + q * 8);
    pa0 = s[0];
    pa1 = s[1];
  };
  auto commitA = [&](unsigned* La) {
    u32x4 vv;
    vv[0] = pack2(pa0.x, pa0.y); vv[1] = pack2(pa0.z, pa0.w);
    vv[2] = pack2(pa1.x, pa1.y); vv[3] = pack2(pa1.z, pa1.w);
    *(u32x4*)(La + r * 16 + q * 4) = vv;
  };
  auto fetchB = [&](int kc) {
    const u32x4* s =
        (const u32x4*)(Wt + (size_t)(n0 + nB) * H + kc * 32 + halfB * 16);
    pb0 = s[0];
    pb1 = s[1];
  };
  auto commitB = [&](unsigned* Lb) {
    u32x4* d = (u32x4*)(Lb + nB * 16 + halfB * 8);
    d[0] = pb0;
    d[1] = pb1;
  };

  v8f acc[4];
#pragma unroll
  for (int i = 0; i < 4; ++i) acc[i] = vzero8();

  fetchA(0); fetchB(0);
  commitA(ldsA[0]); commitB(ldsB[0]);
  fetchA(1); fetchB(1);
  __syncthreads();
  for (int kc = 0; kc < 4; ++kc) {
    v16bf a = readA_lds(ldsA[kc & 1], 16, mt, 0, lane);
#pragma unroll
    for (int tt = 0; tt < 4; ++tt) {
      v16bf bb = readB_lds(ldsB[kc & 1], nh * 4 + tt, lane);
      acc[tt] = wmma_bf16(a, bb, acc[tt]);
    }
    if (kc + 1 < 4) { commitA(ldsA[(kc + 1) & 1]); commitB(ldsB[(kc + 1) & 1]); }
    if (kc + 2 < 4) { fetchA(kc + 2); fetchB(kc + 2); }
    __syncthreads();
  }
  int mbase = mt * 16 + ((lane & 16) ? 8 : 0);
#pragma unroll
  for (int tt = 0; tt < 4; ++tt) {
    int nl = (nh * 4 + tt) * 16 + (lane & 15);
    float bn = bias[n0 + nl];
#pragma unroll
    for (int rr = 0; rr < 8; ++rr)
      out[(m0 + mbase + rr) * CODE + n0 + nl] = acc[tt][rr] + bn;
  }
}

// ---------------------------------------------------------------------------
extern "C" void kernel_launch(void* const* d_in, const int* in_sizes, int n_in,
                              void* d_out, int out_size, void* d_ws,
                              size_t ws_size, hipStream_t stream) {
  (void)n_in; (void)out_size; (void)ws_size;
  const float* y      = (const float*)d_in[0];
  const float* coords = (const float*)d_in[1];
  const int*   ei     = (const int*)d_in[2];
  const float* W_in   = (const float*)d_in[3];
  const float* b_in   = (const float*)d_in[4];
  const float* ln_g   = (const float*)d_in[5];
  const float* ln_b   = (const float*)d_in[6];
  const float* eW1    = (const float*)d_in[7];
  const float* eb1    = (const float*)d_in[8];
  const float* eW2    = (const float*)d_in[9];
  const float* eb2    = (const float*)d_in[10];
  const float* nW1    = (const float*)d_in[11];
  const float* nb1    = (const float*)d_in[12];
  const float* nW2    = (const float*)d_in[13];
  const float* nb2    = (const float*)d_in[14];
  const float* W_out  = (const float*)d_in[15];
  const float* b_out  = (const float*)d_in[16];

  const int Nn = in_sizes[1] / 3;  // 16384 nodes
  const int E  = in_sizes[2] / 2;  // ~324K edges

  char* p = (char*)d_ws;
  auto carve = [&](size_t bytes) {
    void* r = (void*)p;
    p += (bytes + 255) & ~(size_t)255;
    return r;
  };
  float* h      = (float*)carve((size_t)Nn * H * 4);
  float* hn     = (float*)carve((size_t)Nn * H * 4);
  unsigned short* hnb = (unsigned short*)carve((size_t)Nn * H * 2);
  float* agg    = (float*)carve((size_t)Nn * H * 4);
  float* invdeg = (float*)carve((size_t)Nn * 4);
  float* dist   = (float*)carve((size_t)E * 4);
  unsigned short* Wt_in  = (unsigned short*)carve((size_t)H * CODE * 2);
  unsigned short* Wt_e1  = (unsigned short*)carve((size_t)NLAYER * H * 288 * 2);
  unsigned short* Wt_e2  = (unsigned short*)carve((size_t)NLAYER * H * H * 2);
  unsigned short* Wt_n1  = (unsigned short*)carve((size_t)NLAYER * H * 256 * 2);
  unsigned short* Wt_n2  = (unsigned short*)carve((size_t)NLAYER * H * H * 2);
  unsigned short* Wt_out = (unsigned short*)carve((size_t)CODE * H * 2);

  // weight prep (transpose + bf16 + pad)
  wprep_kernel<<<128, 256, 0, stream>>>(W_in, Wt_in, CODE, H, CODE);
  for (int l = 0; l < NLAYER; ++l) {
    wprep_kernel<<<64, 256, 0, stream>>>(eW1 + (size_t)l * (2 * H + 1) * H,
                                         Wt_e1 + (size_t)l * H * 288,
                                         2 * H + 1, H, 288);
    wprep_kernel<<<32, 256, 0, stream>>>(eW2 + (size_t)l * H * H,
                                         Wt_e2 + (size_t)l * H * H, H, H, H);
    wprep_kernel<<<64, 256, 0, stream>>>(nW1 + (size_t)l * 2 * H * H,
                                         Wt_n1 + (size_t)l * H * 2 * H, 2 * H,
                                         H, 2 * H);
    wprep_kernel<<<32, 256, 0, stream>>>(nW2 + (size_t)l * H * H,
                                         Wt_n2 + (size_t)l * H * H, H, H, H);
  }
  wprep_kernel<<<128, 256, 0, stream>>>(W_out, Wt_out, H, CODE, H);

  // graph prep: dist + 1/deg
  zero_kernel<<<(Nn + 255) / 256, 256, 0, stream>>>(invdeg, Nn);
  edge_prep_kernel<<<(E + 255) / 256, 256, 0, stream>>>(coords, ei, E, dist,
                                                        invdeg);
  deg_inv_kernel<<<(Nn + 255) / 256, 256, 0, stream>>>(invdeg, Nn);

  // input projection
  gemm_in_kernel<<<Nn / 64, 256, 0, stream>>>(y, Wt_in, b_in, h);

  for (int l = 0; l < NLAYER; ++l) {
    ln_kernel<<<Nn / 8, 256, 0, stream>>>(h, ln_g + (size_t)l * H,
                                          ln_b + (size_t)l * H, hn, hnb);
    zero_kernel<<<2048, 256, 0, stream>>>(agg, Nn * H);
    edge_kernel<<<(E + 63) / 64, 256, 0, stream>>>(
        ei, E, hnb, dist, Wt_e1 + (size_t)l * H * 288, eb1 + (size_t)l * H,
        Wt_e2 + (size_t)l * H * H, eb2 + (size_t)l * H, agg);
    node_kernel<<<Nn / 64, 256, 0, stream>>>(
        hn, hnb, agg, invdeg, Wt_n1 + (size_t)l * H * 2 * H,
        nb1 + (size_t)l * H, Wt_n2 + (size_t)l * H * H, nb2 + (size_t)l * H, h);
  }

  gemm_out_kernel<<<dim3(Nn / 64, CODE / 128), 256, 0, stream>>>(
      h, Wt_out, b_out, (float*)d_out);
}
